// GraphClassificationNet_9852654977189
// MI455X (gfx1250) — compile-verified, matrix operand order checked by reading, not run on hardware
//
#include <hip/hip_runtime.h>
#include <hip/hip_bf16.h>
#include <math.h>

#define KS 5
#define NN 60000
#define EE 360000
#define N1C 30000
#define E2C 180000
#define N2C 15000
#define GC 60
#define F1 32
#define F2 64
#define ZS 26              /* 25 spline slots + 1 root slot */
#define ZC (ZS * F2)       /* 1664 columns */
#define NT (ZC / 16)       /* 104 n-tiles  */
#define MT (N1C / 16)      /* 1875 m-tiles */
#define NTCHUNK 52         /* n-tiles staged in LDS per chunk: 52*512 bf16 = 53,248 B */

typedef __attribute__((ext_vector_type(16))) __bf16 v16bf;
typedef __attribute__((ext_vector_type(8)))  float  v8f;

union V16 { v16bf v; uint4 q[2]; unsigned short s[16]; };

__device__ __forceinline__ unsigned short f2bf(float f) {
    unsigned u = __float_as_uint(f);
    return (unsigned short)((u + 0x7FFFu + ((u >> 16) & 1u)) >> 16);  // RNE
}
__device__ __forceinline__ float bf2f(unsigned short s) {
    return __uint_as_float(((unsigned)s) << 16);
}
// Monotone float<->uint encoding so atomicMax(uint) == float max (handles negatives).
__device__ __forceinline__ unsigned f2ord(float f) {
    unsigned u = __float_as_uint(f);
    return (u & 0x80000000u) ? ~u : (u | 0x80000000u);
}
__device__ __forceinline__ float ord2f(unsigned o) {
    unsigned u = (o & 0x80000000u) ? (o & 0x7FFFFFFFu) : ~o;
    return __uint_as_float(u);
}
__device__ __forceinline__ float eluf(float x) { return x > 0.f ? x : expm1f(x); }

// Open B-spline basis, degree 1, dim 2: 4 (index, weight) taps per edge.
__device__ __forceinline__ void spline_basis(float ax, float ay, int* idx, float* w) {
    float vx = ax * (KS - 1), vy = ay * (KS - 1);
    float flx = floorf(vx), fly = floorf(vy);
    int ix0 = min(max((int)flx, 0), KS - 1);
    int iy0 = min(max((int)fly, 0), KS - 1);
    float fx = vx - flx, fy = vy - fly;
#pragma unroll
    for (int s0 = 0; s0 < 2; ++s0)
#pragma unroll
        for (int s1 = 0; s1 < 2; ++s1) {
            int ix = min(ix0 + s0, KS - 1);
            int iy = min(iy0 + s1, KS - 1);
            idx[s0 * 2 + s1] = ix + KS * iy;
            w[s0 * 2 + s1] = (s0 ? fx : 1.f - fx) * (s1 ? fy : 1.f - fy);
        }
}

// ---------------- conv1: Fin=1 -> 32, scatter messages via atomics ----------------
__global__ void k_conv1_edge(const float* __restrict__ x, const float* __restrict__ eattr,
                             const int* __restrict__ eidx, const float* __restrict__ W1,
                             float* __restrict__ agg1, float* __restrict__ cnt1) {
    int gid = blockIdx.x * blockDim.x + threadIdx.x;
    if (gid >= EE * F1) return;
    int e = gid >> 5, f = gid & 31;
    int src = eidx[e], dst = eidx[EE + e];
    int idx[4]; float w[4];
    spline_basis(eattr[2 * e], eattr[2 * e + 1], idx, w);
    float wk = 0.f;
#pragma unroll
    for (int t = 0; t < 4; ++t) wk += w[t] * W1[idx[t] * F1 + f];
    atomicAdd(&agg1[dst * F1 + f], x[src] * wk);
    if (f == 0) atomicAdd(&cnt1[dst], 1.f);
}

__global__ void k_conv1_node(const float* __restrict__ x, const float* __restrict__ agg1,
                             const float* __restrict__ cnt1, const float* __restrict__ root1,
                             const float* __restrict__ bias1, float* __restrict__ h1) {
    int gid = blockIdx.x * blockDim.x + threadIdx.x;
    if (gid >= NN * F1) return;
    int n = gid >> 5, f = gid & 31;
    h1[gid] = eluf(agg1[gid] / fmaxf(cnt1[n], 1.f) + x[n] * root1[f] + bias1[f]);
}

// ---------------- pool1: segment max (ordinal atomics) + mean pos + batch max ----------------
__global__ void k_pool1_scatter(const float* __restrict__ h1, const float* __restrict__ pos,
                                const int* __restrict__ batch, const int* __restrict__ cluster1,
                                unsigned* __restrict__ x1ord, float* __restrict__ possum,
                                float* __restrict__ pcnt, int* __restrict__ batch1) {
    int gid = blockIdx.x * blockDim.x + threadIdx.x;
    if (gid >= NN * F1) return;
    int n = gid >> 5, f = gid & 31;
    int c = cluster1[n];
    atomicMax(&x1ord[c * F1 + f], f2ord(h1[gid]));
    if (f < 2) atomicAdd(&possum[c * 2 + f], pos[n * 2 + f]);
    if (f == 0) { atomicAdd(&pcnt[c], 1.f); atomicMax(&batch1[c], batch[n]); }
}

__global__ void k_pool1_final(const unsigned* __restrict__ x1ord, const float* __restrict__ possum,
                              const float* __restrict__ pcnt, unsigned short* __restrict__ a1bf,
                              float* __restrict__ pos1) {
    int gid = blockIdx.x * blockDim.x + threadIdx.x;
    if (gid >= N1C * F1) return;
    int c = gid >> 5, f = gid & 31;
    unsigned o = x1ord[gid];
    a1bf[gid] = f2bf(o ? ord2f(o) : 0.f);               // empty cluster -> 0
    if (f < 2) pos1[c * 2 + f] = possum[c * 2 + f] / fmaxf(pcnt[c], 1.f);
}

// ---------------- build WMMA-B operand: [32 x 1664] = [W2 slots | root2], lane-swizzled bf16 ----
__global__ void k_prep_B(const float* __restrict__ W2, const float* __restrict__ root2,
                         unsigned short* __restrict__ bswz) {
    int tid = blockIdx.x * blockDim.x + threadIdx.x;
    if (tid >= NT * 512) return;
    int j = tid & 15, lane = (tid >> 4) & 31, ntile = tid >> 9;
    int half = lane >> 4;
    int k = (j < 8) ? (8 * half + j) : (16 + 8 * half + (j - 8));   // ISA 16-bit B layout
    int col = ntile * 16 + (lane & 15);
    float v;
    if (col < KS * KS * F2) { int s = col >> 6, f = col & 63; v = W2[(s * F1 + k) * F2 + f]; }
    else                    { v = root2[k * F2 + (col - KS * KS * F2)]; }
    bswz[tid] = f2bf(v);
}

// ---------------- global max|cart| over pooled edges (bits of non-neg float are monotone) -------
__global__ void k_scale(const float* __restrict__ pos1, const int* __restrict__ eidx2,
                        unsigned* __restrict__ scaleBits) {
    __shared__ float red[256];
    int gid = blockIdx.x * 256 + threadIdx.x;
    float m = 0.f;
    if (gid < E2C) {
        int s = eidx2[gid], d = eidx2[E2C + gid];
        m = fmaxf(fabsf(pos1[d * 2] - pos1[s * 2]), fabsf(pos1[d * 2 + 1] - pos1[s * 2 + 1]));
    }
    red[threadIdx.x] = m;
    __syncthreads();
    for (int off = 128; off > 0; off >>= 1) {
        if (threadIdx.x < off) red[threadIdx.x] = fmaxf(red[threadIdx.x], red[threadIdx.x + off]);
        __syncthreads();
    }
    if (threadIdx.x == 0) atomicMax(scaleBits, __float_as_uint(red[0]));
}

// ---------------- WMMA GEMM: Z[30000,1664] = X1_bf16[30000,32] x B_bf16[32,1664] ----------------
// Block = 8 waves = 8 m-tiles. B operand staged into LDS in two 53KB chunks; A held in
// registers for the whole 104-tile n-sweep -> 104 WMMAs per wave, K=32 = 1 WMMA per tile.
__global__ void __launch_bounds__(256) k_gemm(const unsigned short* __restrict__ a1bf,
                                              const unsigned short* __restrict__ bswz,
                                              unsigned short* __restrict__ Z) {
    __shared__ unsigned short sB[NTCHUNK * 512];          // 53,248 B
    const int wid  = threadIdx.x >> 5;
    const int lane = threadIdx.x & 31;
    const int mtile = blockIdx.x * 8 + wid;               // wave-uniform validity
    const int half = lane >> 4;

    V16 a;
    if (mtile < MT) {
        const int m = mtile * 16 + (lane & 15);
        const uint4* arow = (const uint4*)(a1bf + m * F1); // row = 64B = 4x uint4
        a.q[0] = arow[half];        // A[m, 8h .. 8h+7]
        a.q[1] = arow[2 + half];    // A[m, 16+8h .. 16+8h+7]
    }

    for (int cb = 0; cb < 2; ++cb) {
        __syncthreads();
        // cooperative stage: 3328 uint4 (= 53,248 B) spread over 256 threads
        for (int i = threadIdx.x; i < NTCHUNK * 512 / 8; i += 256)
            ((uint4*)sB)[i] = ((const uint4*)bswz)[cb * (NTCHUNK * 512 / 8) + i];
        __syncthreads();

        if (mtile < MT) {
            const int mbase = mtile * 16 + 8 * half;
#pragma unroll 4
            for (int nt = 0; nt < NTCHUNK; ++nt) {
                V16 b;
                const uint4* bl = (const uint4*)(sB + (nt * 32 + lane) * 16);
                b.q[0] = bl[0]; b.q[1] = bl[1];
                v8f c = {};
                c = __builtin_amdgcn_wmma_f32_16x16x32_bf16(false, a.v, false, b.v,
                                                            (short)0, c, false, false);
                const int n = (cb * NTCHUNK + nt) * 16 + (lane & 15);
#pragma unroll
                for (int r = 0; r < 8; ++r)
                    Z[(mbase + r) * ZC + n] = f2bf(c[r]); // D: M = 8*half + r, N = lane&15
            }
        }
    }
}

// ---------------- conv2 edge phase: 4-tap gather of Z rows, atomic scatter -----------------
__global__ void k_conv2_edge(const float* __restrict__ pos1, const int* __restrict__ eidx2,
                             const unsigned* __restrict__ scaleBits, const unsigned short* __restrict__ Z,
                             float* __restrict__ agg2, float* __restrict__ cnt2) {
    int gid = blockIdx.x * blockDim.x + threadIdx.x;
    if (gid >= E2C * F2) return;
    int e = gid >> 6, f = gid & 63;
    int s = eidx2[e], d = eidx2[E2C + e];
    float sc = fmaxf(__uint_as_float(*scaleBits), 1e-12f);
    float ax = (pos1[d * 2]     - pos1[s * 2])     / (2.f * sc) + 0.5f;
    float ay = (pos1[d * 2 + 1] - pos1[s * 2 + 1]) / (2.f * sc) + 0.5f;
    int idx[4]; float w[4];
    spline_basis(ax, ay, idx, w);
    const unsigned short* zrow = Z + s * ZC;
    float msg = 0.f;
#pragma unroll
    for (int t = 0; t < 4; ++t) msg += w[t] * bf2f(zrow[idx[t] * F2 + f]);
    atomicAdd(&agg2[d * F2 + f], msg);
    if (f == 0) atomicAdd(&cnt2[d], 1.f);
}

__global__ void k_conv2_node(const float* __restrict__ agg2, const float* __restrict__ cnt2,
                             const unsigned short* __restrict__ Z, const float* __restrict__ bias2,
                             float* __restrict__ h2) {
    int gid = blockIdx.x * blockDim.x + threadIdx.x;
    if (gid >= N1C * F2) return;
    int n = gid >> 6, f = gid & 63;
    float root = bf2f(Z[n * ZC + KS * KS * F2 + f]);   // root term = slot 25 of the GEMM
    h2[gid] = eluf(agg2[gid] / fmaxf(cnt2[n], 1.f) + root + bias2[f]);
}

// ---------------- pool2 + global mean pool ----------------
__global__ void k_pool2_scatter(const float* __restrict__ h2, const int* __restrict__ cluster2,
                                const int* __restrict__ batch1, unsigned* __restrict__ x2ord,
                                int* __restrict__ batch2) {
    int gid = blockIdx.x * blockDim.x + threadIdx.x;
    if (gid >= N1C * F2) return;
    int n = gid >> 6, f = gid & 63;
    int c = cluster2[n];
    atomicMax(&x2ord[c * F2 + f], f2ord(h2[gid]));
    if (f == 0) atomicMax(&batch2[c], batch1[n]);
}

__global__ void k_pool2_global(const unsigned* __restrict__ x2ord, const int* __restrict__ batch2,
                               float* __restrict__ gsum, float* __restrict__ gcnt) {
    int gid = blockIdx.x * blockDim.x + threadIdx.x;
    if (gid >= N2C * F2) return;
    int c = gid >> 6, f = gid & 63;
    unsigned o = x2ord[gid];
    int g = batch2[c];
    atomicAdd(&gsum[g * F2 + f], o ? ord2f(o) : 0.f);
    if (f == 0) atomicAdd(&gcnt[g], 1.f);
}

// ---------------- head: 64->128 ELU, 128->3, log_softmax; one block per graph ----------------
__global__ void k_head(const float* __restrict__ gsum, const float* __restrict__ gcnt,
                       const float* __restrict__ fc1w, const float* __restrict__ fc1b,
                       const float* __restrict__ fc2w, const float* __restrict__ fc2b,
                       float* __restrict__ out) {
    __shared__ float gm[F2], h3[128], lg[3];
    int g = blockIdx.x, t = threadIdx.x;
    if (t < F2) gm[t] = gsum[g * F2 + t] / fmaxf(gcnt[g], 1.f);
    __syncthreads();
    float acc = fc1b[t];
    for (int k = 0; k < F2; ++k) acc += gm[k] * fc1w[k * 128 + t];
    h3[t] = eluf(acc);
    __syncthreads();
    if (t < 3) {
        float l = fc2b[t];
        for (int k = 0; k < 128; ++k) l += h3[k] * fc2w[k * 3 + t];
        lg[t] = l;
    }
    __syncthreads();
    if (t < 3) {
        float mx = fmaxf(lg[0], fmaxf(lg[1], lg[2]));
        float lse = logf(expf(lg[0] - mx) + expf(lg[1] - mx) + expf(lg[2] - mx));
        out[g * 3 + t] = lg[t] - mx - lse;
    }
}

extern "C" void kernel_launch(void* const* d_in, const int* in_sizes, int n_in,
                              void* d_out, int out_size, void* d_ws, size_t ws_size,
                              hipStream_t stream) {
    const float* x        = (const float*)d_in[0];
    const float* pos      = (const float*)d_in[1];
    const float* eattr    = (const float*)d_in[2];
    const int*   eidx     = (const int*)d_in[3];
    const int*   batch    = (const int*)d_in[4];
    const int*   cluster1 = (const int*)d_in[5];
    const int*   eidx2    = (const int*)d_in[6];
    const int*   cluster2 = (const int*)d_in[7];
    const float* W1       = (const float*)d_in[8];
    const float* root1    = (const float*)d_in[9];
    const float* bias1    = (const float*)d_in[10];
    const float* W2       = (const float*)d_in[11];
    const float* root2    = (const float*)d_in[12];
    const float* bias2    = (const float*)d_in[13];
    const float* fc1w     = (const float*)d_in[14];
    const float* fc1b     = (const float*)d_in[15];
    const float* fc2w     = (const float*)d_in[16];
    const float* fc2b     = (const float*)d_in[17];
    float* out = (float*)d_out;

    // ---- carve workspace (256B aligned) ----
    char* w = (char*)d_ws;
    size_t off = 0;
    auto carve = [&](size_t bytes) -> void* {
        void* p = w + off;
        off = (off + bytes + 255) & ~(size_t)255;
        return p;
    };
    float*          agg1      = (float*)carve((size_t)NN * F1 * 4);
    float*          cnt1      = (float*)carve((size_t)NN * 4);
    float*          h1        = (float*)carve((size_t)NN * F1 * 4);
    unsigned*       x1ord     = (unsigned*)carve((size_t)N1C * F1 * 4);
    float*          possum    = (float*)carve((size_t)N1C * 2 * 4);
    float*          pcnt      = (float*)carve((size_t)N1C * 4);
    int*            batch1    = (int*)carve((size_t)N1C * 4);
    unsigned short* a1bf      = (unsigned short*)carve((size_t)N1C * F1 * 2);
    float*          pos1      = (float*)carve((size_t)N1C * 2 * 4);
    unsigned*       scaleBits = (unsigned*)carve(4);
    unsigned short* bswz      = (unsigned short*)carve((size_t)NT * 512 * 2);
    unsigned short* Z         = (unsigned short*)carve((size_t)N1C * ZC * 2);
    float*          agg2      = (float*)carve((size_t)N1C * F2 * 4);
    float*          cnt2      = (float*)carve((size_t)N1C * 4);
    float*          h2        = (float*)carve((size_t)N1C * F2 * 4);
    unsigned*       x2ord     = (unsigned*)carve((size_t)N2C * F2 * 4);
    int*            batch2    = (int*)carve((size_t)N2C * 4);
    float*          gsum      = (float*)carve((size_t)GC * F2 * 4);
    float*          gcnt      = (float*)carve((size_t)GC * 4);

    // ---- zero accumulators / ordinal-max buffers (capture-legal) ----
    hipMemsetAsync(agg1, 0, (size_t)NN * F1 * 4, stream);
    hipMemsetAsync(cnt1, 0, (size_t)NN * 4, stream);
    hipMemsetAsync(x1ord, 0, (size_t)N1C * F1 * 4, stream);
    hipMemsetAsync(possum, 0, (size_t)N1C * 2 * 4, stream);
    hipMemsetAsync(pcnt, 0, (size_t)N1C * 4, stream);
    hipMemsetAsync(batch1, 0, (size_t)N1C * 4, stream);
    hipMemsetAsync(scaleBits, 0, 4, stream);
    hipMemsetAsync(agg2, 0, (size_t)N1C * F2 * 4, stream);
    hipMemsetAsync(cnt2, 0, (size_t)N1C * 4, stream);
    hipMemsetAsync(x2ord, 0, (size_t)N2C * F2 * 4, stream);
    hipMemsetAsync(batch2, 0, (size_t)N2C * 4, stream);
    hipMemsetAsync(gsum, 0, (size_t)GC * F2 * 4, stream);
    hipMemsetAsync(gcnt, 0, (size_t)GC * 4, stream);

    const int B = 256;
    // conv1 + elu
    k_conv1_edge<<<(EE * F1 + B - 1) / B, B, 0, stream>>>(x, eattr, eidx, W1, agg1, cnt1);
    k_conv1_node<<<(NN * F1 + B - 1) / B, B, 0, stream>>>(x, agg1, cnt1, root1, bias1, h1);
    // pool1 (segment max / mean pos / batch max)
    k_pool1_scatter<<<(NN * F1 + B - 1) / B, B, 0, stream>>>(h1, pos, batch, cluster1,
                                                             x1ord, possum, pcnt, batch1);
    k_pool1_final<<<(N1C * F1 + B - 1) / B, B, 0, stream>>>(x1ord, possum, pcnt, a1bf, pos1);
    // conv2 weights -> WMMA-B swizzle; edge-attr scale
    k_prep_B<<<(NT * 512 + B - 1) / B, B, 0, stream>>>(W2, root2, bswz);
    k_scale<<<(E2C + B - 1) / B, B, 0, stream>>>(pos1, eidx2, scaleBits);
    // WMMA GEMM: Z = X1 x [W2|root2]   (8 m-tiles per block, LDS-staged B, 104 WMMAs/wave)
    k_gemm<<<(MT + 7) / 8, B, 0, stream>>>(a1bf, bswz, Z);
    // conv2 edge phase + node phase + elu
    k_conv2_edge<<<(E2C * F2 + B - 1) / B, B, 0, stream>>>(pos1, eidx2, scaleBits, Z, agg2, cnt2);
    k_conv2_node<<<(N1C * F2 + B - 1) / B, B, 0, stream>>>(agg2, cnt2, Z, bias2, h2);
    // pool2 + global mean pool
    k_pool2_scatter<<<(N1C * F2 + B - 1) / B, B, 0, stream>>>(h2, cluster2, batch1, x2ord, batch2);
    k_pool2_global<<<(N2C * F2 + B - 1) / B, B, 0, stream>>>(x2ord, batch2, gsum, gcnt);
    // FC head + log_softmax
    k_head<<<GC, 128, 0, stream>>>(gsum, gcnt, fc1w, fc1b, fc2w, fc2b, out);
}
